// GCNBuNNModule_5875515261215
// MI455X (gfx1250) — matrix operand order repaired and checked by reading, not compiled, not run on hardware
//
#include <hip/hip_runtime.h>
#include <hip/hip_bf16.h>
#include <math.h>

typedef __attribute__((ext_vector_type(16))) _Float16 v16h;
typedef __attribute__((ext_vector_type(8)))  _Float16 v8h;
typedef __attribute__((ext_vector_type(8)))  float    v8f;

#define DIM   96
#define HID   192
#define TWO_D 192
// packed-fragment geometry: fragment = 32 lanes x 16 f16 = 512 f16 = 1 KB
#define W1_FRAGS (6 * 12)   // 6 K-chunks x 12 col tiles
#define W2_FRAGS (6 * 6)    // 6 K-chunks x 6  col tiles

// ---------------------------------------------------------------- utilities
__device__ __forceinline__ float gelu_tanh(float v) {
    const float c = 0.7978845608028654f; // sqrt(2/pi)
    float t = tanhf(c * (v + 0.044715f * v * v * v));
    return 0.5f * v * (1.0f + t);
}

// ------------------------------------------------- 1) degree / in-count
__global__ void gcn_count_kernel(const int* __restrict__ src, const int* __restrict__ dst,
                                 unsigned* __restrict__ degS, unsigned* __restrict__ cntD, int E) {
    int e = blockIdx.x * blockDim.x + threadIdx.x;
    if (e < E) {
        atomicAdd(&degS[src[e]], 1u);   // out-degree (reference's deg)
        atomicAdd(&cntD[dst[e]], 1u);   // in-count for CSR-by-dst
    }
}

// ------------------------------------------------- 2) exclusive scan (1 block)
__global__ void gcn_scan_kernel(const unsigned* __restrict__ cnt,
                                unsigned* __restrict__ rowptr, int n) {
    __shared__ unsigned buf[1024];
    __shared__ unsigned running;
    if (threadIdx.x == 0) running = 0u;
    __syncthreads();
    for (int base = 0; base < n; base += 1024) {
        int i = base + (int)threadIdx.x;
        unsigned v = (i < n) ? cnt[i] : 0u;
        buf[threadIdx.x] = v;
        __syncthreads();
        for (int off = 1; off < 1024; off <<= 1) {     // Hillis-Steele inclusive scan
            unsigned t = (threadIdx.x >= (unsigned)off) ? buf[threadIdx.x - off] : 0u;
            __syncthreads();
            buf[threadIdx.x] += t;
            __syncthreads();
        }
        unsigned incl = buf[threadIdx.x];
        if (i < n) rowptr[i] = running + (incl - v);
        __syncthreads();
        if (threadIdx.x == 1023) running += buf[1023];
        __syncthreads();
    }
    if (threadIdx.x == 0) rowptr[n] = running;
}

// ------------------------------------------------- 3) CSR scatter + edge norm
__global__ void gcn_build_csr_kernel(const int* __restrict__ src, const int* __restrict__ dst,
                                     const unsigned* __restrict__ degS,
                                     const unsigned* __restrict__ rowptr,
                                     unsigned* __restrict__ cursor,
                                     unsigned* __restrict__ csrc, float* __restrict__ cnorm, int E) {
    int e = blockIdx.x * blockDim.x + threadIdx.x;
    if (e >= E) return;
    int s = src[e], d = dst[e];
    unsigned pos = rowptr[d] + atomicAdd(&cursor[d], 1u);
    csrc[pos]  = (unsigned)s;
    cnorm[pos] = rsqrtf((float)degS[s] * (float)degS[d]);
}

// ------------------------------------------------- 4) one propagation round
// wave per node, lane = feature (3 floats per lane: d = lane, lane+32, lane+64)
__global__ void gcn_prop_kernel(const float* __restrict__ cin, float* __restrict__ cout,
                                float* __restrict__ hacc,
                                const unsigned* __restrict__ rowptr,
                                const unsigned* __restrict__ csrc,
                                const float* __restrict__ cnorm,
                                float scale, int n) {
    int lane = threadIdx.x & 31;
    int node = blockIdx.x * (blockDim.x >> 5) + (threadIdx.x >> 5);
    if (node >= n) return;
    unsigned beg = rowptr[node], end = rowptr[node + 1];
    float a0 = 0.f, a1 = 0.f, a2 = 0.f;
    for (unsigned j = beg; j < end; ++j) {
        unsigned s = csrc[j];
        float    w = cnorm[j];
        const float* p = cin + (size_t)s * DIM;
        a0 = fmaf(p[lane],      w, a0);
        a1 = fmaf(p[lane + 32], w, a1);
        a2 = fmaf(p[lane + 64], w, a2);
    }
    const float* c = cin + (size_t)node * DIM;
    float n0 = scale * (c[lane]      - a0);
    float n1 = scale * (c[lane + 32] - a1);
    float n2 = scale * (c[lane + 64] - a2);
    float* o = cout + (size_t)node * DIM;
    o[lane] = n0; o[lane + 32] = n1; o[lane + 64] = n2;
    float* hh = hacc + (size_t)node * DIM;
    hh[lane] += n0; hh[lane + 32] += n1; hh[lane + 64] += n2;
}

// ------------------------------------------------- 5a) pack [x||h] -> f16 row-major
__global__ void gcn_pack_xh_kernel(const float* __restrict__ x, const float* __restrict__ h,
                                   _Float16* __restrict__ xh, int n) {
    int i = blockIdx.x * blockDim.x + threadIdx.x;
    if (i >= n * TWO_D) return;
    int row = i / TWO_D;
    int k   = i - row * TWO_D;
    float v = (k < DIM) ? x[(size_t)row * DIM + k] : h[(size_t)row * DIM + (k - DIM)];
    xh[i] = (_Float16)v;
}

// ------------------------------------------------- 5b) pack weights into WMMA
// B-fragment layout: frag(c,t), lane, j -> W[(c*32 + 8*(lane>>4) + (j&7) + 16*(j>>3))][t*16 + (lane&15)]
// so each lane's 16 f16 are contiguous in the packed buffer (2x b128 loads later).
__global__ void gcn_packw_kernel(const float* __restrict__ w1, const float* __restrict__ w2,
                                 _Float16* __restrict__ w1p, _Float16* __restrict__ w2p) {
    int idx = blockIdx.x * blockDim.x + threadIdx.x;
    if (idx < W1_FRAGS * 512) {
        int c    = idx / (12 * 512);
        int rem  = idx % (12 * 512);
        int t    = rem / 512;
        int lane = (rem % 512) / 16;
        int j    = idx & 15;
        int col  = t * 16 + (lane & 15);
        int K    = c * 32 + ((lane >> 4) & 1) * 8 + (j & 7) + (j >> 3) * 16;
        w1p[idx] = (_Float16)w1[(size_t)K * HID + col];
    }
    if (idx < W2_FRAGS * 512) {
        int c    = idx / (6 * 512);
        int rem  = idx % (6 * 512);
        int t    = rem / 512;
        int lane = (rem % 512) / 16;
        int j    = idx & 15;
        int col  = t * 16 + (lane & 15);
        int K    = c * 32 + ((lane >> 4) & 1) * 8 + (j & 7) + (j >> 3) * 16;
        w2p[idx] = (_Float16)w2[(size_t)K * DIM + col];
    }
}

// ------------------------------------------------- 6) fused FFN via WMMA
// 256 threads = 8 waves; packed weights staged once into LDS and shared by all
// waves; each wave owns one 16-row tile. All operand loads are 128-bit.
__global__ void __launch_bounds__(256)
gcn_ffn_wmma_kernel(const _Float16* __restrict__ xh,
                    const _Float16* __restrict__ w1p, const float* __restrict__ b1,
                    const _Float16* __restrict__ w2p, const float* __restrict__ b2,
                    float* __restrict__ out, int nrows) {
    __shared__ _Float16 w1s[W1_FRAGS * 512];   // 72 KB
    __shared__ _Float16 w2s[W2_FRAGS * 512];   // 36 KB
    __shared__ _Float16 hids[8 * 16 * HID];    // 48 KB (per-wave gelu staging)

    const int tid  = threadIdx.x;
    const int lane = tid & 31;
    const int wv   = tid >> 5;
    const int l15  = lane & 15;
    const int half = (lane >> 4) & 1;

    // cooperative b128 staging of packed weights into LDS
    {
        const int4* s1 = (const int4*)w1p;
        int4*       d1 = (int4*)w1s;
        for (int i = tid; i < (W1_FRAGS * 512) / 8; i += 256) d1[i] = s1[i];
        const int4* s2 = (const int4*)w2p;
        int4*       d2 = (int4*)w2s;
        for (int i = tid; i < (W2_FRAGS * 512) / 8; i += 256) d2[i] = s2[i];
    }
    __syncthreads();

    const int ntiles = (nrows + 15) / 16;
    const int mytile = blockIdx.x * 8 + wv;
    const bool active = (mytile < ntiles);
    const int tile = active ? mytile : (ntiles - 1);   // clamp: keep EXEC all-ones
    const int m0 = tile * 16;
    int r = m0 + l15;
    if (r >= nrows) r = nrows - 1;
    const _Float16* arow = xh + (size_t)r * TWO_D;
    _Float16* hid = hids + wv * 16 * HID;

    // ---------------- GEMM1: [16,192] x [192,192]
    v8f acc[12] = {};
#pragma unroll
    for (int c = 0; c < 6; ++c) {
        const int kb = c * 32 + half * 8;
        const v8h alo = *(const v8h*)(arow + kb);        // K kb..kb+7   (b128)
        const v8h ahi = *(const v8h*)(arow + kb + 16);   // K kb+16..+23 (b128)
        v16h a;
#pragma unroll
        for (int i = 0; i < 8; ++i) { a[i] = alo[i]; a[8 + i] = ahi[i]; }
#pragma unroll
        for (int t = 0; t < 12; ++t) {
            const v16h b = *(const v16h*)(w1s + ((c * 12 + t) * 32 + lane) * 16);
            acc[t] = __builtin_amdgcn_wmma_f32_16x16x32_f16(
                false, a, false, b, (short)0, acc[t], false, false);
        }
    }

    // ---------------- bias + gelu -> per-wave LDS tile (C layout -> row-major)
#pragma unroll
    for (int t = 0; t < 12; ++t) {
        const int col  = t * 16 + l15;
        const float bs = b1[col];
#pragma unroll
        for (int i = 0; i < 8; ++i) {
            const int M = i + half * 8;                  // C/D layout: VGPR i -> row M
            hid[M * HID + col] = (_Float16)gelu_tanh(acc[t][i] + bs);
        }
    }
    __syncthreads();

    // ---------------- GEMM2: [16,192] x [192,96]
    v8f acc2[6] = {};
#pragma unroll
    for (int c = 0; c < 6; ++c) {
        const int kb = c * 32 + half * 8;
        const v8h alo = *(const v8h*)(hid + l15 * HID + kb);
        const v8h ahi = *(const v8h*)(hid + l15 * HID + kb + 16);
        v16h a;
#pragma unroll
        for (int i = 0; i < 8; ++i) { a[i] = alo[i]; a[8 + i] = ahi[i]; }
#pragma unroll
        for (int t = 0; t < 6; ++t) {
            const v16h b = *(const v16h*)(w2s + ((c * 6 + t) * 32 + lane) * 16);
            acc2[t] = __builtin_amdgcn_wmma_f32_16x16x32_f16(
                false, a, false, b, (short)0, acc2[t], false, false);
        }
    }

    // ---------------- bias + store
#pragma unroll
    for (int t = 0; t < 6; ++t) {
        const int col  = t * 16 + l15;
        const float bs = b2[col];
#pragma unroll
        for (int i = 0; i < 8; ++i) {
            const int row = m0 + i + half * 8;
            if (active && row < nrows) out[(size_t)row * DIM + col] = acc2[t][i] + bs;
        }
    }
}

// ---------------------------------------------------------------- launcher
extern "C" void kernel_launch(void* const* d_in, const int* in_sizes, int n_in,
                              void* d_out, int out_size, void* d_ws, size_t ws_size,
                              hipStream_t stream) {
    const float* x   = (const float*)d_in[0];
    const int*   src = (const int*)d_in[1];
    const int*   dst = (const int*)d_in[2];
    const float* w1  = (const float*)d_in[3];
    const float* b1  = (const float*)d_in[4];
    const float* w2  = (const float*)d_in[5];
    const float* b2  = (const float*)d_in[6];

    const int N = in_sizes[0] / DIM;
    const int E = in_sizes[1];

    // workspace carve-out (256B aligned)
    char* ws = (char*)d_ws;
    size_t off = 0;
    auto take = [&](size_t bytes) -> char* {
        char* p = ws + off;
        off = (off + bytes + 255) & ~(size_t)255;
        return p;
    };
    unsigned* degS   = (unsigned*)take((size_t)N * 4);
    unsigned* cntD   = (unsigned*)take((size_t)N * 4);
    unsigned* cursor = (unsigned*)take((size_t)N * 4);
    unsigned* rowptr = (unsigned*)take((size_t)(N + 1) * 4);
    unsigned* csrc   = (unsigned*)take((size_t)E * 4);
    float*    cnorm  = (float*)take((size_t)E * 4);
    float*    curA   = (float*)take((size_t)N * DIM * 4);
    float*    curB   = (float*)take((size_t)N * DIM * 4);
    float*    hbuf   = (float*)take((size_t)N * DIM * 4);
    _Float16* xh16   = (_Float16*)take((size_t)N * TWO_D * 2);
    _Float16* w1p    = (_Float16*)take((size_t)W1_FRAGS * 512 * 2);
    _Float16* w2p    = (_Float16*)take((size_t)W2_FRAGS * 512 * 2);

    hipMemsetAsync(degS,   0, (size_t)N * 4, stream);
    hipMemsetAsync(cntD,   0, (size_t)N * 4, stream);
    hipMemsetAsync(cursor, 0, (size_t)N * 4, stream);

    const int TB = 256;
    gcn_count_kernel<<<(E + TB - 1) / TB, TB, 0, stream>>>(src, dst, degS, cntD, E);
    gcn_scan_kernel<<<1, 1024, 0, stream>>>(cntD, rowptr, N);
    gcn_build_csr_kernel<<<(E + TB - 1) / TB, TB, 0, stream>>>(src, dst, degS, rowptr,
                                                               cursor, csrc, cnorm, E);

    hipMemcpyAsync(curA, x, (size_t)N * DIM * 4, hipMemcpyDeviceToDevice, stream);
    hipMemcpyAsync(hbuf, x, (size_t)N * DIM * 4, hipMemcpyDeviceToDevice, stream);

    gcn_packw_kernel<<<(W1_FRAGS * 512 + TB - 1) / TB, TB, 0, stream>>>(w1, w2, w1p, w2p);

    // 8 Laplacian power-series rounds (double-buffered, gather-only, no atomics)
    const float TAU = 0.1f;
    float* cin  = curA;
    float* cout = curB;
    const int WPB = 8;                                  // 8 waves (nodes) per block
    const int gridP = (N + WPB - 1) / WPB;
    for (int k = 1; k <= 8; ++k) {
        gcn_prop_kernel<<<gridP, 32 * WPB, 0, stream>>>(cin, cout, hbuf, rowptr, csrc,
                                                        cnorm, -TAU / (float)k, N);
        float* t = cin; cin = cout; cout = t;
    }

    // pack [x||h] to f16 (A-operand feed), then fused WMMA FFN
    gcn_pack_xh_kernel<<<((size_t)N * TWO_D + TB - 1) / TB, TB, 0, stream>>>(x, hbuf, xh16, N);

    const int ntiles = (N + 15) / 16;
    gcn_ffn_wmma_kernel<<<(ntiles + 7) / 8, 256, 0, stream>>>(xh16, w1p, b1, w2p, b2,
                                                              (float*)d_out, N);
}
